// TTTSequentialModel_58480274702563
// MI455X (gfx1250) — compile-verified, matrix operand order checked by reading, not run on hardware
//
#include <hip/hip_runtime.h>
#include <stdint.h>

// ---------------------------------------------------------------------------
// TTT sequential model for MI455X (gfx1250, wave32, WMMA 16x16x32 f16->f32).
//
// The per-token fast-weight scan is re-expressed exactly as:
//   K = X X^T + 1 (per batch)            -> WMMA GEMM
//   R = X W0^T + b0 - X_next             -> WMMA GEMM
//   D = (I + LR*tril(K,-1))^{-1} R       -> wave-local sliced triangular solve
//   pred = D + X_next                    -> elementwise
//   act  = sigmoid(pred gw^T+gb)*(pred uw^T+ub)   -> WMMA GEMM (dual-B)
//   y    = act dw^T + db + pred ; out = LN(y)     -> WMMA GEMM + LN
// Head: logits = x hw^T + hb              -> WMMA GEMM (dominant 16.8 GFLOP)
// ---------------------------------------------------------------------------

typedef __attribute__((ext_vector_type(4)))  float    vf4;
typedef __attribute__((ext_vector_type(8)))  float    vf8;
typedef __attribute__((ext_vector_type(8)))  float    v8f;
typedef __attribute__((ext_vector_type(4)))  _Float16 v4h;
typedef __attribute__((ext_vector_type(8)))  _Float16 v8h;
typedef __attribute__((ext_vector_type(16))) _Float16 v16h;

#define HD   512      // hidden
#define SD   256      // sequence
#define BD   2        // batch
#define H4   2048     // 4*H
#define VD   32000    // vocab
#define MR   512      // B*S rows
#define LRF  0.01f
#define EPSF 1e-5f

// ---------------- WMMA helpers (ISA 7.12.2 layouts) ----------------

__device__ __forceinline__ v8f wmma16(v16h a, v16h b, v8f c) {
  return __builtin_amdgcn_wmma_f32_16x16x32_f16(false, a, false, b, (short)0, c,
                                                false, false);
}

// A/B fragment from f16 row-major [rows x ld]; lane holds row (base+lane&15),
// K elements (e<8?0:16) + 8*(lane>>4) + (e&7) relative to kk.
__device__ __forceinline__ v16h frag_f16(const _Float16* p, int ld, int row,
                                         int kk, int lane) {
  int r  = row + (lane & 15);
  int kh = (lane >> 4) << 3;
  const _Float16* q = p + (size_t)r * ld + kk + kh;
  v8h lo = *(const v8h*)(q);
  v8h hi = *(const v8h*)(q + 16);
  return __builtin_shufflevector(lo, hi, 0, 1, 2, 3, 4, 5, 6, 7, 8, 9, 10, 11,
                                 12, 13, 14, 15);
}

// Same fragment but sourced from f32 weights, converted on the fly
// (weights are streamed once; avoids a separate conversion pass).
__device__ __forceinline__ v16h frag_f32(const float* p, int ld, int row,
                                         int kk, int lane) {
  int r  = row + (lane & 15);
  int kh = (lane >> 4) << 3;
  const float* q = p + (size_t)r * ld + kk + kh;
  vf8 f0 = *(const vf8*)(q);
  vf8 f1 = *(const vf8*)(q + 16);
  v8h lo = __builtin_convertvector(f0, v8h);
  v8h hi = __builtin_convertvector(f1, v8h);
  return __builtin_shufflevector(lo, hi, 0, 1, 2, 3, 4, 5, 6, 7, 8, 9, 10, 11,
                                 12, 13, 14, 15);
}

// ---------------- small elementwise kernels ----------------

__global__ void embed_kernel(const int* __restrict__ ids,
                             const float* __restrict__ emb,
                             float* __restrict__ X) {
  int r  = blockIdx.x;
  int id = ids[r];
  const vf4* src = (const vf4*)(emb + (size_t)id * HD);
  vf4* dst = (vf4*)(X + (size_t)r * HD);
  dst[threadIdx.x] = src[threadIdx.x];   // 128 thr * 4 = 512 floats
}

__global__ void cvt_kernel(const float* __restrict__ x,
                           _Float16* __restrict__ xh) {
  int i = (blockIdx.x * blockDim.x + threadIdx.x) * 4;  // MR*HD elements
  vf4 v = *(const vf4*)(x + i);
  *(v4h*)(xh + i) = __builtin_convertvector(v, v4h);
}

__global__ void make_pred_kernel(const float* __restrict__ D,
                                 const float* __restrict__ X,
                                 float* __restrict__ P,
                                 _Float16* __restrict__ Ph) {
  int i = (blockIdx.x * blockDim.x + threadIdx.x) * 4;
  int t = i / HD;
  int s = t & (SD - 1);
  vf4 d = *(const vf4*)(D + i);
  vf4 xn = {0.f, 0.f, 0.f, 0.f};
  if (s != SD - 1) xn = *(const vf4*)(X + i + HD);  // next token, same batch
  vf4 p = d + xn;
  *(vf4*)(P + i) = p;
  *(v4h*)(Ph + i) = __builtin_convertvector(p, v4h);
}

__global__ __launch_bounds__(128) void ln_kernel(const float* __restrict__ Y,
                                                 const float* __restrict__ g,
                                                 const float* __restrict__ b,
                                                 float* __restrict__ X) {
  int t = blockIdx.x, tid = threadIdx.x;
  int lane = tid & 31, wid = tid >> 5;
  vf4 v = *(const vf4*)(Y + (size_t)t * HD + tid * 4);
  float sum = v[0] + v[1] + v[2] + v[3];
  float sq  = v[0] * v[0] + v[1] * v[1] + v[2] * v[2] + v[3] * v[3];
  for (int o = 16; o > 0; o >>= 1) {
    sum += __shfl_xor(sum, o, 32);
    sq  += __shfl_xor(sq, o, 32);
  }
  __shared__ float ssum[4], ssq[4];
  if (lane == 0) { ssum[wid] = sum; ssq[wid] = sq; }
  __syncthreads();
  sum = ssum[0] + ssum[1] + ssum[2] + ssum[3];
  sq  = ssq[0] + ssq[1] + ssq[2] + ssq[3];
  float mean = sum * (1.0f / HD);
  float var  = sq * (1.0f / HD) - mean * mean;
  float inv  = rsqrtf(var + EPSF);
  vf4 gg = *(const vf4*)(g + tid * 4);
  vf4 bb = *(const vf4*)(b + tid * 4);
  vf4 o;
  for (int i = 0; i < 4; ++i) o[i] = (v[i] - mean) * inv * gg[i] + bb[i];
  *(vf4*)(X + (size_t)t * HD + tid * 4) = o;
}

// ---------------- triangular solve (only sequential part) ----------------
// D_t[h] = R_t[h] - LR * sum_{s<t} K[t][s] * D_s[h]; independent per (b,h).
//
// 4 lanes cooperate per column (lane = col*4 + slice). All 4 lanes of a
// column live in one wave32 => LDS writes/reads are ordered by DScnt within
// the wave and NO barriers are needed. Partial sums combine with two
// shfl_xor butterflies. Column history is kept in LDS with a +1 stride pad
// so lanes with different s-slices hit different banks:
//   addr = s*65 + col  ->  bank = (s + col) & 63.
#define TSCOL 64   // columns per block (256 threads = 64 cols * 4 slices)
__global__ __launch_bounds__(256) void tri_solve_kernel(
    const float* __restrict__ R, const float* __restrict__ Km,
    float* __restrict__ D) {
  extern __shared__ float sm[];          // [SD][TSCOL+1]
  const int STRIDE = TSCOL + 1;          // 65
  int lane = threadIdx.x & 31, wid = threadIdx.x >> 5;
  int slice = lane & 3;                  // 0..3  (s-strip)
  int col   = wid * 8 + (lane >> 2);     // 0..63 (column within block)
  const int nb = HD / TSCOL;             // 8 column-blocks per batch
  int bb = blockIdx.x / nb;
  int h  = (blockIdx.x % nb) * TSCOL + col;
  const float* Kb = Km + (size_t)bb * SD * SD;
  const float* Rb = R + (size_t)bb * SD * HD;
  float* Db = D + (size_t)bb * SD * HD;
  for (int t = 0; t < SD; ++t) {
    const float* Krow = Kb + (size_t)t * SD;
    float acc = 0.f;
#pragma unroll 4
    for (int s = slice; s < t; s += 4)
      acc += Krow[s] * sm[s * STRIDE + col];
    acc += __shfl_xor(acc, 1, 32);       // combine the 4 slices
    acc += __shfl_xor(acc, 2, 32);
    float d = Rb[(size_t)t * HD + h] - LRF * acc;
    if (slice == 0) {
      sm[t * STRIDE + col] = d;
      Db[(size_t)t * HD + h] = d;
    }
  }
}

// ---------------- WMMA GEMM kernels ----------------
// Block: 256 thr = 8 waves in 2(M)x4(N); wave tile 32x32 = 2x2 WMMA frags.
// Block tile 64x128; all dims divide evenly -> EXEC all ones, no masking.

#define GEMM_PROLOG()                                   \
  int lane = threadIdx.x & 31, wid = threadIdx.x >> 5;  \
  int wm = wid & 1, wn = wid >> 1;                      \
  int m0 = blockIdx.x * 64 + wm * 32;                   \
  int n0 = blockIdx.y * 128 + wn * 32;                  \
  v8f acc[2][2] = {};

// R = Xh * W0^T + b0 - X_next
__global__ __launch_bounds__(256) void gemm_R_kernel(
    const _Float16* __restrict__ Ah, const float* __restrict__ W0,
    const float* __restrict__ b0, const float* __restrict__ X,
    float* __restrict__ R) {
  GEMM_PROLOG();
  for (int kk = 0; kk < HD; kk += 32) {
    v16h a0 = frag_f16(Ah, HD, m0, kk, lane);
    v16h a1 = frag_f16(Ah, HD, m0 + 16, kk, lane);
    v16h bb0 = frag_f32(W0, HD, n0, kk, lane);
    v16h bb1 = frag_f32(W0, HD, n0 + 16, kk, lane);
    acc[0][0] = wmma16(a0, bb0, acc[0][0]);
    acc[0][1] = wmma16(a0, bb1, acc[0][1]);
    acc[1][0] = wmma16(a1, bb0, acc[1][0]);
    acc[1][1] = wmma16(a1, bb1, acc[1][1]);
  }
  for (int mt = 0; mt < 2; ++mt)
    for (int nt = 0; nt < 2; ++nt) {
      int n = n0 + nt * 16 + (lane & 15);
      int mb = m0 + mt * 16 + ((lane >> 4) << 3);
      float bias = b0[n];
      for (int r = 0; r < 8; ++r) {
        int m = mb + r;
        int s = m & (SD - 1);
        float xn = (s == SD - 1) ? 0.f : X[(size_t)(m + 1) * HD + n];
        R[(size_t)m * HD + n] = acc[mt][nt][r] + bias - xn;
      }
    }
}

// K[b] = Xb Xb^T + 1   (per batch via blockIdx.z)
__global__ __launch_bounds__(256) void gemm_K_kernel(
    const _Float16* __restrict__ Xh, float* __restrict__ Km) {
  GEMM_PROLOG();
  const _Float16* Ab = Xh + (size_t)blockIdx.z * SD * HD;
  float* Kb = Km + (size_t)blockIdx.z * SD * SD;
  for (int kk = 0; kk < HD; kk += 32) {
    v16h a0 = frag_f16(Ab, HD, m0, kk, lane);
    v16h a1 = frag_f16(Ab, HD, m0 + 16, kk, lane);
    v16h bb0 = frag_f16(Ab, HD, n0, kk, lane);
    v16h bb1 = frag_f16(Ab, HD, n0 + 16, kk, lane);
    acc[0][0] = wmma16(a0, bb0, acc[0][0]);
    acc[0][1] = wmma16(a0, bb1, acc[0][1]);
    acc[1][0] = wmma16(a1, bb0, acc[1][0]);
    acc[1][1] = wmma16(a1, bb1, acc[1][1]);
  }
  for (int mt = 0; mt < 2; ++mt)
    for (int nt = 0; nt < 2; ++nt) {
      int n = n0 + nt * 16 + (lane & 15);
      int mb = m0 + mt * 16 + ((lane >> 4) << 3);
      for (int r = 0; r < 8; ++r)
        Kb[(size_t)(mb + r) * SD + n] = acc[mt][nt][r] + 1.0f;
    }
}

// act = sigmoid(pred gw^T + gb) * (pred uw^T + ub)  (dual-B accumulation)
__global__ __launch_bounds__(256) void gemm_GU_kernel(
    const _Float16* __restrict__ Ph, const float* __restrict__ GW,
    const float* __restrict__ GB, const float* __restrict__ UW,
    const float* __restrict__ UB, _Float16* __restrict__ ACT) {
  int lane = threadIdx.x & 31, wid = threadIdx.x >> 5;
  int wm = wid & 1, wn = wid >> 1;
  int m0 = blockIdx.x * 64 + wm * 32;
  int n0 = blockIdx.y * 128 + wn * 32;
  v8f ag[2][2] = {}, au[2][2] = {};
  for (int kk = 0; kk < HD; kk += 32) {
    v16h a0 = frag_f16(Ph, HD, m0, kk, lane);
    v16h a1 = frag_f16(Ph, HD, m0 + 16, kk, lane);
    v16h g0 = frag_f32(GW, HD, n0, kk, lane);
    v16h g1 = frag_f32(GW, HD, n0 + 16, kk, lane);
    v16h u0 = frag_f32(UW, HD, n0, kk, lane);
    v16h u1 = frag_f32(UW, HD, n0 + 16, kk, lane);
    ag[0][0] = wmma16(a0, g0, ag[0][0]);
    ag[0][1] = wmma16(a0, g1, ag[0][1]);
    ag[1][0] = wmma16(a1, g0, ag[1][0]);
    ag[1][1] = wmma16(a1, g1, ag[1][1]);
    au[0][0] = wmma16(a0, u0, au[0][0]);
    au[0][1] = wmma16(a0, u1, au[0][1]);
    au[1][0] = wmma16(a1, u0, au[1][0]);
    au[1][1] = wmma16(a1, u1, au[1][1]);
  }
  for (int mt = 0; mt < 2; ++mt)
    for (int nt = 0; nt < 2; ++nt) {
      int n = n0 + nt * 16 + (lane & 15);
      int mb = m0 + mt * 16 + ((lane >> 4) << 3);
      float gbias = GB[n], ubias = UB[n];
      for (int r = 0; r < 8; ++r) {
        float g = ag[mt][nt][r] + gbias;
        float u = au[mt][nt][r] + ubias;
        float sg = 1.0f / (1.0f + __expf(-g));
        ACT[(size_t)(mb + r) * H4 + n] = (_Float16)(sg * u);
      }
    }
}

// Y = act dw^T + db + pred   (K = 2048)
__global__ __launch_bounds__(256) void gemm_down_kernel(
    const _Float16* __restrict__ Ah, const float* __restrict__ DWp,
    const float* __restrict__ DBp, const float* __restrict__ P,
    float* __restrict__ Y) {
  GEMM_PROLOG();
  for (int kk = 0; kk < H4; kk += 32) {
    v16h a0 = frag_f16(Ah, H4, m0, kk, lane);
    v16h a1 = frag_f16(Ah, H4, m0 + 16, kk, lane);
    v16h bb0 = frag_f32(DWp, H4, n0, kk, lane);
    v16h bb1 = frag_f32(DWp, H4, n0 + 16, kk, lane);
    acc[0][0] = wmma16(a0, bb0, acc[0][0]);
    acc[0][1] = wmma16(a0, bb1, acc[0][1]);
    acc[1][0] = wmma16(a1, bb0, acc[1][0]);
    acc[1][1] = wmma16(a1, bb1, acc[1][1]);
  }
  for (int mt = 0; mt < 2; ++mt)
    for (int nt = 0; nt < 2; ++nt) {
      int n = n0 + nt * 16 + (lane & 15);
      int mb = m0 + mt * 16 + ((lane >> 4) << 3);
      float bias = DBp[n];
      for (int r = 0; r < 8; ++r) {
        int m = mb + r;
        Y[(size_t)m * HD + n] = acc[mt][nt][r] + bias + P[(size_t)m * HD + n];
      }
    }
}

// logits = x hw^T + hb   (N = 32000, the dominant GEMM; memory-bound:
// 64MB weights read + 64MB logits written ~ 5.6us at 23.3 TB/s)
__global__ __launch_bounds__(256) void gemm_head_kernel(
    const _Float16* __restrict__ Ah, const float* __restrict__ HWp,
    const float* __restrict__ HBp, float* __restrict__ out) {
  GEMM_PROLOG();
  for (int kk = 0; kk < HD; kk += 32) {
    v16h a0 = frag_f16(Ah, HD, m0, kk, lane);
    v16h a1 = frag_f16(Ah, HD, m0 + 16, kk, lane);
    v16h bb0 = frag_f32(HWp, HD, n0, kk, lane);
    v16h bb1 = frag_f32(HWp, HD, n0 + 16, kk, lane);
    acc[0][0] = wmma16(a0, bb0, acc[0][0]);
    acc[0][1] = wmma16(a0, bb1, acc[0][1]);
    acc[1][0] = wmma16(a1, bb0, acc[1][0]);
    acc[1][1] = wmma16(a1, bb1, acc[1][1]);
  }
  for (int mt = 0; mt < 2; ++mt)
    for (int nt = 0; nt < 2; ++nt) {
      int n = n0 + nt * 16 + (lane & 15);
      int mb = m0 + mt * 16 + ((lane >> 4) << 3);
      float bias = HBp[n];
      for (int r = 0; r < 8; ++r)
        out[(size_t)(mb + r) * VD + n] = acc[mt][nt][r] + bias;
    }
}

// ---------------- host launch ----------------

extern "C" void kernel_launch(void* const* d_in, const int* in_sizes, int n_in,
                              void* d_out, int out_size, void* d_ws,
                              size_t ws_size, hipStream_t stream) {
  (void)in_sizes; (void)n_in; (void)out_size; (void)ws_size;
  const int*   ids = (const int*)d_in[0];
  const float* emb = (const float*)d_in[1];
  const float* W0  = (const float*)d_in[2];
  const float* B0  = (const float*)d_in[3];
  const float* GW  = (const float*)d_in[4];
  const float* GB  = (const float*)d_in[5];
  const float* UW  = (const float*)d_in[6];
  const float* UB  = (const float*)d_in[7];
  const float* DW  = (const float*)d_in[8];
  const float* DB  = (const float*)d_in[9];
  const float* LG  = (const float*)d_in[10];
  const float* LB  = (const float*)d_in[11];
  const float* HW  = (const float*)d_in[12];
  const float* HB  = (const float*)d_in[13];
  float* out = (float*)d_out;

  uint8_t* ws = (uint8_t*)d_ws;
  const size_t MB = (size_t)1 << 20;
  float*    X  = (float*)(ws + 0 * MB);   // [MR,HD] activations
  float*    Rb = (float*)(ws + 1 * MB);   // [MR,HD]
  float*    Km = (float*)(ws + 2 * MB);   // [BD,SD,SD]
  float*    Db = (float*)(ws + 3 * MB);   // [MR,HD]
  float*    Pr = (float*)(ws + 4 * MB);   // [MR,HD] pred f32
  float*    Yb = (float*)(ws + 5 * MB);   // [MR,HD]
  _Float16* Xh = (_Float16*)(ws + 6 * MB);  // [MR,HD] f16
  _Float16* Ph = (_Float16*)(ws + 7 * MB);  // [MR,HD] f16
  _Float16* Ah = (_Float16*)(ws + 8 * MB);  // [MR,H4] f16 (2MB)

  embed_kernel<<<MR, 128, 0, stream>>>(ids, emb, X);

  for (int l = 0; l < 2; ++l) {
    const float* w0 = W0 + (size_t)l * HD * HD;
    const float* b0 = B0 + (size_t)l * HD;
    const float* gw = GW + (size_t)l * H4 * HD;
    const float* gb = GB + (size_t)l * H4;
    const float* uw = UW + (size_t)l * H4 * HD;
    const float* ub = UB + (size_t)l * H4;
    const float* dw = DW + (size_t)l * HD * H4;
    const float* db = DB + (size_t)l * HD;
    const float* lg = LG + (size_t)l * HD;
    const float* lb = LB + (size_t)l * HD;

    cvt_kernel<<<256, 256, 0, stream>>>(X, Xh);
    gemm_R_kernel<<<dim3(MR / 64, HD / 128), 256, 0, stream>>>(Xh, w0, b0, X, Rb);
    gemm_K_kernel<<<dim3(SD / 64, SD / 128, BD), 256, 0, stream>>>(Xh, Km);
    tri_solve_kernel<<<BD * (HD / TSCOL), 256,
                       SD * (TSCOL + 1) * sizeof(float), stream>>>(Rb, Km, Db);
    make_pred_kernel<<<256, 256, 0, stream>>>(Db, X, Pr, Ph);
    gemm_GU_kernel<<<dim3(MR / 64, H4 / 128), 256, 0, stream>>>(Ph, gw, gb, uw,
                                                                ub, Ah);
    gemm_down_kernel<<<dim3(MR / 64, HD / 128), 256, 0, stream>>>(Ah, dw, db,
                                                                  Pr, Yb);
    ln_kernel<<<MR, 128, 0, stream>>>(Yb, lg, lb, X);
  }

  cvt_kernel<<<256, 256, 0, stream>>>(X, Xh);
  gemm_head_kernel<<<dim3(MR / 64, VD / 128), 256, 0, stream>>>(Xh, HW, HB, out);
}
